// VectorQuantizer_30142080483334
// MI455X (gfx1250) — matrix operand order repaired and checked by reading, not compiled, not run on hardware
//
#include <hip/hip_runtime.h>

typedef __attribute__((ext_vector_type(16))) __bf16 v16bf;
typedef __attribute__((ext_vector_type(8)))  float  v8f;

#define N_TOK   16384
#define N_EMB   256
#define N_CODE  8192
#define TTILES  (N_TOK / 16)     // 1024 token tiles
#define CTILES  (N_CODE / 16)    // 512 code tiles
#define KCH     (N_EMB / 32)     // 8 K-chunks of 32
#define SLICES  2                // code slices per block
#define TT_BLK  4                // token tiles per block
#define CT_PER_SLICE (CTILES / SLICES)  // 256
#define TILE_BYTES 8192          // one packed bf16 B tile (hi or lo): 8*32*16*2
#define BUF_BYTES  (2 * TILE_BYTES)     // hi + lo = 16 KB

// ---------------------------------------------------------------------------
// Pack X into WMMA A-fragment layout (16-bit A 16x32, ISA 7.12.2), bf16 hi/lo.
// Linear packed index t = ((tt*KCH + kc)*32 + lane)*16 + i
// ---------------------------------------------------------------------------
__global__ void __launch_bounds__(256) vq_prep_a(const float* __restrict__ X,
                                                 __bf16* __restrict__ Ah,
                                                 __bf16* __restrict__ Al) {
    unsigned t    = blockIdx.x * 256u + threadIdx.x;
    unsigned i    = t & 15u;
    unsigned lane = (t >> 4) & 31u;
    unsigned kc   = (t >> 9) & 7u;
    unsigned tt   = t >> 12;
    unsigned m = lane & 15u, h = lane >> 4;
    unsigned v = i >> 1, e = i & 1u;
    unsigned Kl = (v < 4u ? 2u * v + e : 16u + 2u * (v - 4u) + e) + h * 8u;
    unsigned k  = kc * 32u + Kl;
    float x = X[(size_t)(tt * 16u + m) * N_EMB + k];
    __bf16 hi = (__bf16)x;
    float  lo = x - (float)hi;
    Ah[t] = hi;
    Al[t] = (__bf16)lo;
}

// ---------------------------------------------------------------------------
// Pack E^T into WMMA B-fragment layout (16-bit B 32x16): lanes 0-15 hold
// K=0..7,16..23, lanes 16-31 hold K=8..15,24..31; lane%16 = column N = code.
// ---------------------------------------------------------------------------
__global__ void __launch_bounds__(256) vq_prep_b(const float* __restrict__ E,
                                                 __bf16* __restrict__ Bh,
                                                 __bf16* __restrict__ Bl) {
    unsigned t    = blockIdx.x * 256u + threadIdx.x;
    unsigned i    = t & 15u;
    unsigned lane = (t >> 4) & 31u;
    unsigned kc   = (t >> 9) & 7u;
    unsigned ct   = t >> 12;
    unsigned n = lane & 15u, h = lane >> 4;
    unsigned Kl = (i < 8u ? i : 8u + i) + h * 8u;
    unsigned k  = kc * 32u + Kl;
    unsigned code = ct * 16u + n;
    float x = E[(size_t)code * N_EMB + k];
    __bf16 hi = (__bf16)x;
    float  lo = x - (float)hi;
    Bh[t] = hi;
    Bl[t] = (__bf16)lo;
}

// ---------------------------------------------------------------------------
// bias[c] = 0.5 * ||e_c||^2   (argmin dist == argmax (x.e - bias))
// ---------------------------------------------------------------------------
__global__ void __launch_bounds__(256) vq_prep_bias(const float* __restrict__ E,
                                                    float* __restrict__ bias) {
    int c = blockIdx.x * 256 + threadIdx.x;
    const float4* p = (const float4*)(E + (size_t)c * N_EMB);
    float s = 0.f;
    #pragma unroll 4
    for (int q = 0; q < N_EMB / 4; ++q) {
        float4 v = p[q];
        s += v.x * v.x + v.y * v.y + v.z * v.z + v.w * v.w;
    }
    bias[c] = 0.5f * s;
}

// ---------------------------------------------------------------------------
// Async copy 16 bytes global -> LDS (gfx1250, tracked by ASYNCcnt).
// ---------------------------------------------------------------------------
__device__ __forceinline__ void async_copy16(unsigned lds_off, const void* gbase,
                                             unsigned goff) {
    asm volatile("global_load_async_to_lds_b128 %0, %1, %2 offset:0"
                 :: "v"(lds_off), "v"(goff), "s"(gbase) : "memory");
}
__device__ __forceinline__ void wait_async0() {
    asm volatile("s_wait_asynccnt 0x0" ::: "memory");
}

// ---------------------------------------------------------------------------
// Main fused GEMM + argmax + gather.
// Block = 256 threads = 8 waves = 4 token tiles x 2 code slices.
// The 4 waves of a slice share each 16 KB B tile via double-buffered LDS,
// staged with async-to-LDS copies that overlap the 24 bf16 WMMAs per tile.
// ---------------------------------------------------------------------------
__global__ void __launch_bounds__(256) vq_main(const __bf16* __restrict__ Ah,
                                               const __bf16* __restrict__ Al,
                                               const __bf16* __restrict__ Bh,
                                               const __bf16* __restrict__ Bl,
                                               const float*  __restrict__ bias,
                                               const float*  __restrict__ E,
                                               float* __restrict__ out) {
    __shared__ __attribute__((aligned(32))) char smem[2 * 2 * BUF_BYTES]; // 64 KB

    const int lane  = threadIdx.x & 31;
    const int wib   = threadIdx.x >> 5;     // 0..7
    const int slice = wib & 1;              // code slice
    const int tq    = wib >> 1;             // token tile within block (0..3)
    const int tile  = blockIdx.x * TT_BLK + tq;
    const int ncol  = lane & 15;
    const int h     = lane >> 4;
    const int sid   = tq * 32 + lane;       // 0..127: stage id within slice group

    const unsigned smem_base  = (unsigned)(size_t)(&smem[0]);
    const unsigned slice_base = smem_base + (unsigned)slice * (2u * BUF_BYTES);

    // Resident A fragments: 16 tokens x 256 K, bf16 hi+lo (128 VGPRs)
    v16bf ah[KCH], al[KCH];
    #pragma unroll
    for (int kc = 0; kc < KCH; ++kc) {
        ah[kc] = *(const v16bf*)(Ah + (((size_t)tile * KCH + kc) * 32 + lane) * 16);
        al[kc] = *(const v16bf*)(Al + (((size_t)tile * KCH + kc) * 32 + lane) * 16);
    }

    float bestv[8];
    int   besti[8];
    #pragma unroll
    for (int j = 0; j < 8; ++j) { bestv[j] = -3.402823466e38f; besti[j] = 0; }

    // Prologue: stage first tile of this slice into buffer 0.
    {
        const unsigned ct0 = (unsigned)slice;  // ct = 2*ts + slice, ts = 0
        #pragma unroll
        for (int q = 0; q < 4; ++q) {
            unsigned c = (unsigned)sid + (unsigned)q * 128u;
            async_copy16(slice_base + c * 16u,              Bh, ct0 * TILE_BYTES + c * 16u);
            async_copy16(slice_base + TILE_BYTES + c * 16u, Bl, ct0 * TILE_BYTES + c * 16u);
        }
    }
    wait_async0();
    __syncthreads();

    for (int ts = 0; ts < CT_PER_SLICE; ++ts) {
        const int ct = 2 * ts + slice;
        // Stage next tile into the other buffer (overlaps compute below).
        if (ts + 1 < CT_PER_SLICE) {
            const unsigned nct  = (unsigned)(2 * (ts + 1) + slice);
            const unsigned nbuf = slice_base + (unsigned)((ts + 1) & 1) * BUF_BYTES;
            #pragma unroll
            for (int q = 0; q < 4; ++q) {
                unsigned c = (unsigned)sid + (unsigned)q * 128u;
                async_copy16(nbuf + c * 16u,              Bh, nct * TILE_BYTES + c * 16u);
                async_copy16(nbuf + TILE_BYTES + c * 16u, Bl, nct * TILE_BYTES + c * 16u);
            }
        }

        const float b = bias[ct * 16 + ncol];
        const char* buf = smem + (size_t)slice * (2 * BUF_BYTES) + (size_t)(ts & 1) * BUF_BYTES;

        v8f a0 = {}, a1 = {}, a2 = {};
        #pragma unroll
        for (int kc = 0; kc < KCH; ++kc) {
            v16bf bh = *(const v16bf*)(buf + (kc * 32 + lane) * 32);
            v16bf bl = *(const v16bf*)(buf + TILE_BYTES + (kc * 32 + lane) * 32);
            a0 = __builtin_amdgcn_wmma_f32_16x16x32_bf16(false, ah[kc], false, bh,
                                                         (short)0, a0, false, false);
            a1 = __builtin_amdgcn_wmma_f32_16x16x32_bf16(false, al[kc], false, bh,
                                                         (short)0, a1, false, false);
            a2 = __builtin_amdgcn_wmma_f32_16x16x32_bf16(false, ah[kc], false, bl,
                                                         (short)0, a2, false, false);
        }

        const int code = ct * 16 + ncol;
        #pragma unroll
        for (int j = 0; j < 8; ++j) {
            float v = (a0[j] + a1[j] + a2[j]) - b;   // x.e - 0.5||e||^2, maximize
            if (v > bestv[j]) { bestv[j] = v; besti[j] = code; }
        }

        wait_async0();      // our async stores to the other buffer done
        __syncthreads();    // whole block in lockstep; safe to reuse buffers
    }

    // Butterfly argmax across the 16 lanes of each half (tile columns).
    #pragma unroll
    for (int m = 1; m < 16; m <<= 1) {
        #pragma unroll
        for (int j = 0; j < 8; ++j) {
            float ov = __shfl_xor(bestv[j], m, 32);
            int   oi = __shfl_xor(besti[j], m, 32);
            if (ov > bestv[j] || (ov == bestv[j] && oi < besti[j])) {
                bestv[j] = ov; besti[j] = oi;
            }
        }
    }

    // Merge scratch aliases slice0/buf0 (dead: final iteration read buf 1).
    float* s_val = (float*)smem;            // [TT_BLK*16][SLICES]
    int*   s_idx = (int*)(smem + 2048);

    if (ncol == 0) {
        #pragma unroll
        for (int j = 0; j < 8; ++j) {
            const int row = tq * 16 + h * 8 + j;
            s_val[row * SLICES + slice] = bestv[j];
            s_idx[row * SLICES + slice] = besti[j];
        }
    }
    __syncthreads();

    // Each wave merges the 16 rows of its token tile (lane = row), gathers 8.
    int mi = 0;
    if (lane < 16) {
        const int base = (tq * 16 + lane) * SLICES;
        float mv = s_val[base];
        mi = s_idx[base];
        float v  = s_val[base + 1];
        int   id = s_idx[base + 1];
        if (v > mv || (v == mv && id < mi)) { mv = v; mi = id; }
    }

    #pragma unroll
    for (int j = 0; j < 8; ++j) {
        const int row  = slice * 8 + j;
        const int code = __shfl(mi, row, 32);
        const float4* src = (const float4*)(E + (size_t)code * N_EMB);
        float4*       dst = (float4*)(out + ((size_t)tile * 16 + row) * N_EMB);
        dst[lane]      = src[lane];        // 64 float4 per row, 32 lanes x 2
        dst[lane + 32] = src[lane + 32];
    }
}

// ---------------------------------------------------------------------------
extern "C" void kernel_launch(void* const* d_in, const int* in_sizes, int n_in,
                              void* d_out, int out_size, void* d_ws, size_t ws_size,
                              hipStream_t stream) {
    const float* X = (const float*)d_in[0];   // [8,2048,256] fp32
    const float* E = (const float*)d_in[1];   // [8192,256]  fp32
    float* out = (float*)d_out;

    char* ws = (char*)d_ws;
    const size_t nA = (size_t)TTILES * KCH * 32 * 16;   // 4,194,304 bf16
    const size_t nB = (size_t)CTILES * KCH * 32 * 16;   // 2,097,152 bf16
    __bf16* Ah = (__bf16*)ws;
    __bf16* Al = Ah + nA;
    __bf16* Bh = Al + nA;
    __bf16* Bl = Bh + nB;
    float*  bias = (float*)(Bl + nB);                   // total ~25.2 MB of ws

    vq_prep_a   <<<(unsigned)(nA / 256), 256, 0, stream>>>(X, Ah, Al);
    vq_prep_b   <<<(unsigned)(nB / 256), 256, 0, stream>>>(E, Bh, Bl);
    vq_prep_bias<<<N_CODE / 256,        256, 0, stream>>>(E, bias);
    vq_main     <<<TTILES / TT_BLK,     256, 0, stream>>>(Ah, Al, Bh, Bl, bias, E, out);
}
